// HMMTrajNet_57501022159262
// MI455X (gfx1250) — matrix operand order repaired and checked by reading, not compiled, not run on hardware
//
#include <hip/hip_runtime.h>
#include <hip/hip_bf16.h>

#define T_LEN  8192
#define NROWS  8193        // T+1 rows of s_i
#define SDIM   256
#define NCOLS  4288        // 4096 action | 128 stop | 64 start
#define NCT    268         // NCOLS / 16 column tiles
#define PW_DW  (NCT * 2048) // == 128 * NCOLS bf16-pair dwords, tile-major layout
#define ROW64  (NROWS * 64)

typedef __attribute__((ext_vector_type(16))) __bf16 v16bf;
typedef __attribute__((ext_vector_type(8)))  float  v8f;

union Frag { unsigned u[8]; v16bf v; };

__device__ __forceinline__ unsigned f2bf1(float f) {
  unsigned u = __float_as_uint(f);
  return (u + 0x7FFFu + ((u >> 16) & 1u)) >> 16;   // round-to-nearest-even bf16
}
__device__ __forceinline__ unsigned pack2(float lo, float hi) {
  return f2bf1(lo) | (f2bf1(hi) << 16);
}

#define WMMA_BF16(A, B, C) \
  __builtin_amdgcn_wmma_f32_16x16x32_bf16(false, (A), false, (B), (short)0, (C), false, false)

// ---------------------------------------------------------------------------
// Kernel 0: pack W_action|W_stop|W_start into tile-major bf16 B-fragments.
// Layout: pw[(((ct*8 + kc)*2 + hi)*16 + n)*8 + v] holds the bf16 pair
// (W[k][c], W[k+1][c]) with c = ct*16+n, k = kc*32 + hi*8 + 2*(v&3) + 16*(v>>2),
// i.e. each lane's 8 fragment dwords are contiguous -> 2x b128 per WMMA.
// ---------------------------------------------------------------------------
__global__ __launch_bounds__(256)
void pack_weights(const float* __restrict__ Wa, const float* __restrict__ Ws,
                  const float* __restrict__ Wst, unsigned* __restrict__ pw) {
  int idx = blockIdx.x * 256 + threadIdx.x;
  if (idx >= PW_DW) return;
  const int v  = idx & 7;
  const int n  = (idx >> 3) & 15;
  const int hi = (idx >> 7) & 1;
  const int kc = (idx >> 8) & 7;
  const int ct = idx >> 11;
  const int c  = ct * 16 + n;
  const int s0 = kc * 32 + hi * 8 + (v & 3) * 2 + (v >> 2) * 16;    // even K of the pair
  const int s1 = s0 + 1;
  float f0, f1;
  if (c < 4096)      { f0 = Wa[s0 * 4096 + c];          f1 = Wa[s1 * 4096 + c]; }
  else if (c < 4224) { int cc = c - 4096; f0 = Ws[s0 * 128 + cc];  f1 = Ws[s1 * 128 + cc]; }
  else               { int cc = c - 4224; f0 = Wst[s0 * 64 + cc];  f1 = Wst[s1 * 64 + cc]; }
  pw[idx] = pack2(f0, f1);
}

// ---------------------------------------------------------------------------
// Kernel 1: WMMA GEMM (16 t-rows per WG, 4 waves) + fused log-softmax outputs:
//   emit[t][b], beta[t][b]=stop_logps[t,b,0], omb[t][b]=stop_logps[t,b,1],
//   start[t][b]=start_logps[t,b]
// ---------------------------------------------------------------------------
__global__ __launch_bounds__(128)
void hmm_gemm_softmax(const float* __restrict__ s_i,
                      const int* __restrict__ actions,
                      const unsigned* __restrict__ pw,
                      float* __restrict__ emitO, float* __restrict__ betaO,
                      float* __restrict__ ombO,  float* __restrict__ startO) {
  __shared__ float lds[4][16 * 64];
  const int lane = threadIdx.x & 31;
  const int wave = threadIdx.x >> 5;
  const int hi   = lane >> 4;      // lane half
  const int n    = lane & 15;
  const int t0   = blockIdx.x * 16;
  float* blk = lds[wave];

  // ---- Preload A fragments: 16 rows x 256 K, register-resident, ISA layout:
  // VGPR v(0..3): K = 2v(+8 for hi half); v(4..7): K = 16+2(v-4)(+8); chunks of 32.
  unsigned afr[8][8];
  {
    const int  row = t0 + n;
    const bool ok  = row < NROWS;
    const float* sp = s_i + row * SDIM;
    #pragma unroll
    for (int kc = 0; kc < 8; ++kc) {
      #pragma unroll
      for (int v = 0; v < 8; ++v) {
        const int k = kc * 32 + (v & 3) * 2 + ((v >> 2) ? 16 : 0) + hi * 8;
        float x0 = 0.f, x1 = 0.f;
        if (ok) { x0 = sp[k]; x1 = sp[k + 1]; }
        afr[kc][v] = pack2(x0, x1);
      }
    }
  }

  const v8f vzero = {0.f, 0.f, 0.f, 0.f, 0.f, 0.f, 0.f, 0.f};
  // per-lane base into the tile-major packed weights (dword units)
  const int laneBase = (hi * 16 + n) * 8;

  auto loadB = [&](int ct, int kc, Frag& b) {
    const uint4* p =
        (const uint4*)(pw + (ct * 8 + kc) * 256 + laneBase);
    const uint4 q0 = p[0], q1 = p[1];           // 2x global_load_b128
    b.u[0] = q0.x; b.u[1] = q0.y; b.u[2] = q0.z; b.u[3] = q0.w;
    b.u[4] = q1.x; b.u[5] = q1.y; b.u[6] = q1.z; b.u[7] = q1.w;
  };
  auto getA = [&](int kc, Frag& a) {
    #pragma unroll
    for (int v = 0; v < 8; ++v) a.u[v] = afr[kc][v];
  };

  // 64-column block (4 tiles): ping/pong B batches, sched_barriers keep a full
  // 8-load batch in flight across the other batch's 4 WMMAs.
  auto doBlock64 = [&](int ct0, int kind, int bIdx) {
    v8f acc0 = vzero, acc1 = vzero, acc2 = vzero, acc3 = vzero;
    Frag pA0, pA1, pA2, pA3;                    // ping: even kc
    Frag pB0, pB1, pB2, pB3;                    // pong: odd kc
    loadB(ct0 + 0, 0, pA0); loadB(ct0 + 1, 0, pA1);
    loadB(ct0 + 2, 0, pA2); loadB(ct0 + 3, 0, pA3);
    loadB(ct0 + 0, 1, pB0); loadB(ct0 + 1, 1, pB1);
    loadB(ct0 + 2, 1, pB2); loadB(ct0 + 3, 1, pB3);
    __builtin_amdgcn_sched_barrier(0);
    #pragma unroll
    for (int kc = 0; kc < 8; kc += 2) {
      {
        Frag a;  getA(kc, a);
        acc0 = WMMA_BF16(a.v, pA0.v, acc0);
        acc1 = WMMA_BF16(a.v, pA1.v, acc1);
        acc2 = WMMA_BF16(a.v, pA2.v, acc2);
        acc3 = WMMA_BF16(a.v, pA3.v, acc3);
      }
      if (kc + 2 < 8) {                         // refill ping while pong computes
        loadB(ct0 + 0, kc + 2, pA0); loadB(ct0 + 1, kc + 2, pA1);
        loadB(ct0 + 2, kc + 2, pA2); loadB(ct0 + 3, kc + 2, pA3);
      }
      __builtin_amdgcn_sched_barrier(0);
      {
        Frag a;  getA(kc + 1, a);
        acc0 = WMMA_BF16(a.v, pB0.v, acc0);
        acc1 = WMMA_BF16(a.v, pB1.v, acc1);
        acc2 = WMMA_BF16(a.v, pB2.v, acc2);
        acc3 = WMMA_BF16(a.v, pB3.v, acc3);
      }
      if (kc + 3 < 8) {                         // refill pong while ping computes
        loadB(ct0 + 0, kc + 3, pB0); loadB(ct0 + 1, kc + 3, pB1);
        loadB(ct0 + 2, kc + 3, pB2); loadB(ct0 + 3, kc + 3, pB3);
      }
      __builtin_amdgcn_sched_barrier(0);
    }
    #pragma unroll
    for (int r = 0; r < 8; ++r) {              // C layout: M = r + 8*hi, N = n
      const int M = r + hi * 8;
      blk[M * 64 +      n] = acc0[r];
      blk[M * 64 + 16 + n] = acc1[r];
      blk[M * 64 + 32 + n] = acc2[r];
      blk[M * 64 + 48 + n] = acc3[r];
    }
    asm volatile("s_wait_dscnt 0" ::: "memory");
    const int row = n;                          // each half-wave scans 32 cols
    const float* brow = blk + row * 64;
    float mx = -3.4e38f;
    #pragma unroll
    for (int c = 0; c < 32; ++c) mx = fmaxf(mx, brow[hi * 32 + c]);
    mx = fmaxf(mx, __shfl_xor(mx, 16, 32));
    float s = 0.f;
    #pragma unroll
    for (int c = 0; c < 32; ++c) s += __expf(brow[hi * 32 + c] - mx);
    s += __shfl_xor(s, 16, 32);
    const float lse = mx + __logf(s);
    const int t = t0 + row;
    if (kind == 0) {                            // action block -> emit gather
      if (hi == 0 && t < T_LEN) {
        const int a = actions[t];
        emitO[t * 64 + bIdx] = brow[a] - lse;
      }
    } else {                                    // start block -> full row
      if (t < NROWS) {
        #pragma unroll
        for (int c = 0; c < 32; ++c)
          startO[t * 64 + hi * 32 + c] = brow[hi * 32 + c] - lse;
      }
    }
  };

  // Two stop tiles at once, same ping/pong structure.
  auto doStopPair = [&](int st) {               // st even, handles st and st+1
    v8f acc0 = vzero, acc1 = vzero;
    Frag pA0, pA1, pB0, pB1;
    loadB(256 + st, 0, pA0); loadB(256 + st + 1, 0, pA1);
    loadB(256 + st, 1, pB0); loadB(256 + st + 1, 1, pB1);
    __builtin_amdgcn_sched_barrier(0);
    #pragma unroll
    for (int kc = 0; kc < 8; kc += 2) {
      {
        Frag a;  getA(kc, a);
        acc0 = WMMA_BF16(a.v, pA0.v, acc0);
        acc1 = WMMA_BF16(a.v, pA1.v, acc1);
      }
      if (kc + 2 < 8) { loadB(256 + st, kc + 2, pA0); loadB(256 + st + 1, kc + 2, pA1); }
      __builtin_amdgcn_sched_barrier(0);
      {
        Frag a;  getA(kc + 1, a);
        acc0 = WMMA_BF16(a.v, pB0.v, acc0);
        acc1 = WMMA_BF16(a.v, pB1.v, acc1);
      }
      if (kc + 3 < 8) { loadB(256 + st, kc + 3, pB0); loadB(256 + st + 1, kc + 3, pB1); }
      __builtin_amdgcn_sched_barrier(0);
    }
    #pragma unroll
    for (int r = 0; r < 8; ++r) {
      const int M = r + hi * 8;
      blk[M * 64 +      n] = acc0[r];           // tile st   -> cols 0..15
      blk[M * 64 + 16 + n] = acc1[r];           // tile st+1 -> cols 16..31
    }
    asm volatile("s_wait_dscnt 0" ::: "memory");
    #pragma unroll
    for (int rep = 0; rep < 8; ++rep) {
      const int idx = rep * 32 + lane;          // 256 (row, pair) tasks
      const int row = idx >> 4, p = idx & 15;   // p: 0..7 tile st, 8..15 tile st+1
      const int tt = p >> 3, pp = p & 7;
      const float l0 = blk[row * 64 + tt * 16 + 2 * pp];
      const float l1 = blk[row * 64 + tt * 16 + 2 * pp + 1];
      const float m  = fmaxf(l0, l1);
      const float lse = m + __logf(__expf(l0 - m) + __expf(l1 - m));
      const int t = t0 + row, b = (st + tt) * 8 + pp;
      if (t < NROWS) {
        betaO[t * 64 + b] = l0 - lse;           // STOP_IX = 0
        ombO [t * 64 + b] = l1 - lse;           // CONT_IX = 1
      }
    }
  };

  // Wave work split keeps every softmax group inside one wave.
  const int bStart = wave * 16;
  for (int b = bStart; b < bStart + 16; ++b)
    doBlock64(b * 4, 0, b);                     // action block b -> tiles b*4..b*4+3
  if (wave == 0) {
    doBlock64(264, 1, 0);                       // start block (cols 4224..4287)
  } else if (wave == 1) {
    doStopPair(0); doStopPair(2);
  } else if (wave == 2) {
    doStopPair(4); doStopPair(6);
  }
}

// ---------------------------------------------------------------------------
// Kernel 2: serial HMM scan, single wave32; rank-1 + diagonal factorization:
//   S = lse_i(f[i]+beta[i]);  f[j] = logaddexp(S+start[j], f[j]+omb[j]) + em[j]
// ---------------------------------------------------------------------------
__global__ __launch_bounds__(32)
void hmm_scan(const float* __restrict__ emitA, const float* __restrict__ betaA,
              const float* __restrict__ ombA,  const float* __restrict__ startA,
              float* __restrict__ out) {
  const int j = threadIdx.x;                    // lane handles states j, j+32
  float f0 = startA[j]      + emitA[j];
  float f1 = startA[j + 32] + emitA[j + 32];

  // software pipeline: current regs + loads for t+1 in flight
  int o = 64;
  float b0 = betaA[o + j],  b1 = betaA[o + j + 32];
  float s0 = startA[o + j], s1 = startA[o + j + 32];
  float c0 = ombA[o + j],   c1 = ombA[o + j + 32];
  float e0 = emitA[o + j],  e1 = emitA[o + j + 32];

  for (int t = 1; t < T_LEN; ++t) {
    const int on = (t + 1) * 64;                // next-step loads (overlap chain)
    const float nb0 = betaA[on + j],  nb1 = betaA[on + j + 32];
    const float ns0 = startA[on + j], ns1 = startA[on + j + 32];
    const float nc0 = ombA[on + j],   nc1 = ombA[on + j + 32];
    const float ne0 = emitA[on + j],  ne1 = emitA[on + j + 32];
    if (t + 8 < T_LEN) {                        // keep the chain in WGP$/L2
      const int op = (t + 8) * 64 + j;
      __builtin_prefetch(&betaA[op], 0, 0);
      __builtin_prefetch(&startA[op], 0, 0);
      __builtin_prefetch(&ombA[op], 0, 0);
      __builtin_prefetch(&emitA[op], 0, 0);
    }

    float a0 = f0 + b0, a1 = f1 + b1;
    float m = fmaxf(a0, a1);
    #pragma unroll
    for (int off = 16; off; off >>= 1) m = fmaxf(m, __shfl_xor(m, off, 32));
    float sum = __expf(a0 - m) + __expf(a1 - m);
    #pragma unroll
    for (int off = 16; off; off >>= 1) sum += __shfl_xor(sum, off, 32);
    const float S = m + __logf(sum);

    float x = S + s0, y = f0 + c0, mm = fmaxf(x, y);
    f0 = mm + __logf(__expf(x - mm) + __expf(y - mm)) + e0;
    x = S + s1; y = f1 + c1; mm = fmaxf(x, y);
    f1 = mm + __logf(__expf(x - mm) + __expf(y - mm)) + e1;

    b0 = nb0; b1 = nb1; s0 = ns0; s1 = ns1;
    c0 = nc0; c1 = nc1; e0 = ne0; e1 = ne1;
  }

  // total_logp = lse_j(f[j] + stop_logps[T, j, STOP]); return negative
  const int of = T_LEN * 64;
  float a0 = f0 + betaA[of + j], a1 = f1 + betaA[of + j + 32];
  float m = fmaxf(a0, a1);
  #pragma unroll
  for (int off = 16; off; off >>= 1) m = fmaxf(m, __shfl_xor(m, off, 32));
  float sum = __expf(a0 - m) + __expf(a1 - m);
  #pragma unroll
  for (int off = 16; off; off >>= 1) sum += __shfl_xor(sum, off, 32);
  if (j == 0) out[0] = -(m + __logf(sum));
}

// ---------------------------------------------------------------------------
extern "C" void kernel_launch(void* const* d_in, const int* in_sizes, int n_in,
                              void* d_out, int out_size, void* d_ws, size_t ws_size,
                              hipStream_t stream) {
  const float* s_i     = (const float*)d_in[0];
  const float* Wa      = (const float*)d_in[1];
  const float* Ws      = (const float*)d_in[2];
  const float* Wst     = (const float*)d_in[3];
  const int*   actions = (const int*)d_in[4];
  // d_in[5] = lengths (fixed T=8192), unused

  unsigned* pw     = (unsigned*)d_ws;
  float*    base   = (float*)d_ws;
  float*    emitO  = base + PW_DW;
  float*    betaO  = emitO + ROW64;
  float*    ombO   = betaO + ROW64;
  float*    startO = ombO + ROW64;
  // total workspace: (PW_DW + 4*ROW64)*4 B ~= 10.6 MB

  hipLaunchKernelGGL(pack_weights, dim3((PW_DW + 255) / 256), dim3(256), 0, stream,
                     Wa, Ws, Wst, pw);
  hipLaunchKernelGGL(hmm_gemm_softmax, dim3((NROWS + 15) / 16), dim3(128), 0, stream,
                     s_i, actions, pw, emitO, betaO, ombO, startO);
  hipLaunchKernelGGL(hmm_scan, dim3(1), dim3(32), 0, stream,
                     emitO, betaO, ombO, startO, (float*)d_out);
}
